// CompressiveMemory_31507880084133
// MI455X (gfx1250) — matrix operand order repaired and checked by reading, not compiled, not run on hardware
//
#include <hip/hip_runtime.h>

typedef __attribute__((ext_vector_type(16))) __bf16 v16bf;
typedef __attribute__((ext_vector_type(8)))  __bf16 v8bf;
typedef __attribute__((ext_vector_type(8)))  float  v8f;

#define NB 8
#define NH 12
#define NS 4096
#define ND 64
#define SUB 64            // S rows per sub-tile
#define SPW 512           // S rows per workgroup
#define NSUB (SPW / SUB)

// ---------------------------------------------------------------------------
// WMMA helper: D = A(16x32 bf16) * B(32x16 bf16) + C(f32), wave32.
// ---------------------------------------------------------------------------
__device__ __forceinline__ v8f wmma_bf16(v16bf a, v16bf b, v8f c) {
  return __builtin_amdgcn_wmma_f32_16x16x32_bf16(
      /*neg_a=*/false, a, /*neg_b=*/false, b,
      /*c_mod=*/(short)0, c, /*reuse_a=*/false, /*reuse_b=*/false);
}

// A-fragment per ISA 7.12.2 (16-bit A 16x32): lane L holds row m=L%16.
// half = L/16 selects K sub-bands: elems 0..7 <- K = kc+8h .. kc+8h+7,
// elems 8..15 <- K = kc+16+8h .. kc+16+8h+7 (two contiguous 16B segments).
__device__ __forceinline__ v16bf load_a_frag(const __bf16* base, int row,
                                             int stride, int kc, int half) {
  union { v16bf v; v8bf h[2]; } f;
  const __bf16* p = base + row * stride + kc + 8 * half;
  f.h[0] = *(const v8bf*)p;
  f.h[1] = *(const v8bf*)(p + 16);
  return f.v;
}

// B-fragment (32x16): lane L holds column n=col; elems 0..7 <- K = kc+8h..kc+8h+7,
// elems 8..15 <- K = kc+16+8h..kc+16+8h+7 (same K order as A).
__device__ __forceinline__ v16bf load_b_frag(const __bf16* base, int stride,
                                             int kc, int col, int half) {
  v16bf f;
  #pragma unroll
  for (int s = 0; s < 8; ++s) {
    f[s]     = base[(kc + 8 * half + s) * stride + col];
    f[s + 8] = base[(kc + 16 + 8 * half + s) * stride + col];
  }
  return f;
}

__device__ __forceinline__ float act_elu1(float x) {
  // elu(x)+1 == x+1 (x>0) else exp(x)
  return x > 0.f ? x + 1.f : __expf(x);
}

// Async copy 16B/lane from global (saddr + 32-bit voffset) into LDS.
// gfx1250 GLOBAL_LOAD_ASYNC_TO_LDS_B128, tracked by ASYNCcnt.
__device__ __forceinline__ void async_copy_b128(uint32_t lds_byte_off,
                                                const float* base,
                                                uint32_t glb_byte_off) {
  asm volatile("global_load_async_to_lds_b128 %0, %1, %2 offset:0"
               :
               : "v"(lds_byte_off), "v"(glb_byte_off), "s"(base)
               : "memory");
}

__device__ __forceinline__ void wait_asynccnt0() {
#if __has_builtin(__builtin_amdgcn_s_wait_asynccnt)
  __builtin_amdgcn_s_wait_asynccnt(0);
#else
  asm volatile("s_wait_asynccnt 0x0" ::: "memory");
#endif
}

// ---------------------------------------------------------------------------
// Main kernel: one WG per (b, h, s-block of 512 rows).
// ---------------------------------------------------------------------------
__global__ __launch_bounds__(256) void cm_main_kernel(
    const float* __restrict__ Q, const float* __restrict__ K,
    const float* __restrict__ V, const float* __restrict__ Mg,
    const float* __restrict__ zg, float* __restrict__ outA,
    float* __restrict__ dM_acc, float* __restrict__ z_acc) {
  __shared__ alignas(16) __bf16 sM[ND * ND];     // M[h] in bf16, row-major [d][e]
  __shared__ alignas(16) float  sz[ND];
  __shared__ alignas(16) __bf16 sQa[SUB * ND];   // elu(Q)+1
  __shared__ alignas(16) __bf16 sKa[SUB * ND];   // elu(K)+1
  __shared__ alignas(16) __bf16 sKaT[ND * SUB];  // K_act transposed [d][s]
  __shared__ alignas(16) __bf16 sVe[SUB * ND];   // V_err
  __shared__ alignas(16) float  sVf[SUB * ND];   // V staged via async-to-LDS
  __shared__ alignas(16) float  sdq[SUB];        // 1 / (Q_act.z + eps)
  __shared__ alignas(16) float  sdk[SUB];        // 1 / (K_act.z + eps)

  const int tid = threadIdx.x;
  const int blk = blockIdx.x;
  const int sb  = blk & (NS / SPW - 1);          // 8 s-blocks
  const int h   = (blk >> 3) % NH;
  const int b   = blk / ((NS / SPW) * NH);
  const long base_bh = (long)(b * NH + h) * NS * ND;

  for (int i = tid; i < ND * ND; i += 256) sM[i] = (__bf16)Mg[h * ND * ND + i];
  if (tid < ND) sz[tid] = zg[h * ND + tid];

  const int wave = tid >> 5;
  const int lane = tid & 31;
  const int half = lane >> 4;
  const int l16  = lane & 15;

  // Persistent delta_M accumulators: wave owns 2 of the 16 output 16x16 tiles.
  const int t0 = wave * 2;
  const int t1 = t0 + 1;
  v8f accDM0 = {};
  v8f accDM1 = {};
  float zsum = 0.f;

  for (int st = 0; st < NSUB; ++st) {
    const int s0 = sb * SPW + st * SUB;
    __syncthreads();  // previous iteration's sKaT/sVe/sVf fully consumed

    // ---- Phase 1: async-stage V; load Q/K, apply elu+1, stage bf16 ----
    #pragma unroll
    for (int c = 0; c < 4; ++c) {
      const int f4  = c * 256 + tid;             // 1024 float4's = 64x64 f32
      const int row = f4 >> 4;
      const int col = (f4 & 15) << 2;
      const long g  = base_bh + (long)(s0 + row) * ND + col;

      // V: DMA straight into LDS, overlapped with Q/K VALU work below.
      const uint32_t ldsoff = (uint32_t)(uintptr_t)&sVf[row * ND + col];
      async_copy_b128(ldsoff, V, (uint32_t)(g * sizeof(float)));

      const float4 q = *(const float4*)(Q + g);
      const float4 k = *(const float4*)(K + g);
      const float qa[4] = {act_elu1(q.x), act_elu1(q.y), act_elu1(q.z), act_elu1(q.w)};
      const float ka[4] = {act_elu1(k.x), act_elu1(k.y), act_elu1(k.z), act_elu1(k.w)};
      #pragma unroll
      for (int j = 0; j < 4; ++j) {
        sQa[row * ND + col + j] = (__bf16)qa[j];
        const __bf16 kb = (__bf16)ka[j];
        sKa[row * ND + col + j]     = kb;
        sKaT[(col + j) * SUB + row] = kb;
      }
      if (st + 1 < NSUB) {  // gfx1250 global_prefetch_b8 for next sub-tile
        __builtin_prefetch(Q + g + SUB * ND, 0, 1);
        __builtin_prefetch(K + g + SUB * ND, 0, 1);
        __builtin_prefetch(V + g + SUB * ND, 0, 1);
      }
    }
    wait_asynccnt0();  // V tile resident in LDS
    __syncthreads();

    // ---- Denominators (store reciprocal via v_rcp_f32) + z partials ----
    if (tid < SUB) {
      float s = 0.f;
      for (int d = 0; d < ND; ++d) s += (float)sQa[tid * ND + d] * sz[d];
      sdq[tid] = __builtin_amdgcn_rcpf(s + 1e-8f);
    } else if (tid < 2 * SUB) {
      const int r = tid - SUB;
      float s = 0.f;
      for (int d = 0; d < ND; ++d) s += (float)sKa[r * ND + d] * sz[d];
      sdk[r] = __builtin_amdgcn_rcpf(s + 1e-8f);
    } else if (tid < 2 * SUB + ND) {
      const int d = tid - 2 * SUB;
      float s = 0.f;
      for (int r = 0; r < SUB; ++r) s += (float)sKa[r * ND + d];
      zsum += s;
    }
    __syncthreads();

    // ---- Phase 2: retrieve GEMMs (Q_act@M and K_act@M) via WMMA ----
    // wave -> 16 rows (rowbase) and 2 of the 4 N-tiles (nt0, nt0+1)
    const int rowbase = (wave & 3) * 16;
    const int nt0 = (wave >> 2) * 2;
    const int arow = rowbase + l16;
    v8f accQ0 = {}, accQ1 = {}, accK0 = {}, accK1 = {};
    #pragma unroll
    for (int kc = 0; kc < ND; kc += 32) {
      const v16bf aQ = load_a_frag(sQa, arow, ND, kc, half);
      const v16bf aK = load_a_frag(sKa, arow, ND, kc, half);
      const v16bf b0 = load_b_frag(sM, ND, kc, nt0 * 16 + l16, half);
      const v16bf b1 = load_b_frag(sM, ND, kc, nt0 * 16 + 16 + l16, half);
      accQ0 = wmma_bf16(aQ, b0, accQ0);
      accK0 = wmma_bf16(aK, b0, accK0);
      accQ1 = wmma_bf16(aQ, b1, accQ1);
      accK1 = wmma_bf16(aK, b1, accK1);
    }
    // D layout: lane -> n = l16, vgpr r -> m = rowbase + r + 8*half
    #pragma unroll
    for (int r = 0; r < 8; ++r) {
      const int rl = rowbase + r + 8 * half;
      const float rdq = sdq[rl];
      const float rdk = sdk[rl];
      const long orow = base_bh + (long)(s0 + rl) * ND;
      const int n0 = nt0 * 16 + l16;
      outA[orow + n0]      = accQ0[r] * rdq;
      outA[orow + n0 + 16] = accQ1[r] * rdq;
      sVe[rl * ND + n0]      = (__bf16)(sVf[rl * ND + n0]      - accK0[r] * rdk);
      sVe[rl * ND + n0 + 16] = (__bf16)(sVf[rl * ND + n0 + 16] - accK1[r] * rdk);
    }
    __syncthreads();

    // ---- Phase 3: delta_M partial = K_act^T @ V_err, accumulate in regs ----
    #pragma unroll
    for (int sc = 0; sc < SUB; sc += 32) {
      const v16bf a0 = load_a_frag(sKaT, (t0 >> 2) * 16 + l16, SUB, sc, half);
      const v16bf b0 = load_b_frag(sVe, ND, sc, (t0 & 3) * 16 + l16, half);
      accDM0 = wmma_bf16(a0, b0, accDM0);
      const v16bf a1 = load_a_frag(sKaT, (t1 >> 2) * 16 + l16, SUB, sc, half);
      const v16bf b1 = load_b_frag(sVe, ND, sc, (t1 & 3) * 16 + l16, half);
      accDM1 = wmma_bf16(a1, b1, accDM1);
    }
  }

  // ---- One atomic dump of the (b, s-block) partial into global accumulators ----
  #pragma unroll
  for (int r = 0; r < 8; ++r) {
    const int d0 = (t0 >> 2) * 16 + r + 8 * half;
    const int e0 = (t0 & 3) * 16 + l16;
    atomicAdd(&dM_acc[h * ND * ND + d0 * ND + e0], accDM0[r]);
    const int d1 = (t1 >> 2) * 16 + r + 8 * half;
    const int e1 = (t1 & 3) * 16 + l16;
    atomicAdd(&dM_acc[h * ND * ND + d1 * ND + e1], accDM1[r]);
  }
  if (tid >= 2 * SUB && tid < 2 * SUB + ND) {
    atomicAdd(&z_acc[h * ND + (tid - 2 * SUB)], zsum);
  }
}

// ---------------------------------------------------------------------------
__global__ void cm_zero_kernel(float* __restrict__ p, int n) {
  const int i = blockIdx.x * 256 + threadIdx.x;
  if (i < n) p[i] = 0.f;
}

__global__ void cm_finalize_kernel(const float* __restrict__ Mg,
                                   const float* __restrict__ zg,
                                   const float* __restrict__ dM,
                                   const float* __restrict__ dz,
                                   float* __restrict__ outM,
                                   float* __restrict__ outz) {
  const float scale = 0.1f / (float)NB;  // RATE * (1/B)
  const int i = blockIdx.x * 256 + threadIdx.x;
  if (i < NH * ND * ND) {
    outM[i] = (Mg[i] + dM[i] * scale) * 0.95f;
  } else {
    const int j = i - NH * ND * ND;
    if (j < NH * ND) outz[j] = (zg[j] + dz[j] * scale) * 0.95f;
  }
}

// ---------------------------------------------------------------------------
extern "C" void kernel_launch(void* const* d_in, const int* in_sizes, int n_in,
                              void* d_out, int out_size, void* d_ws,
                              size_t ws_size, hipStream_t stream) {
  (void)in_sizes; (void)n_in; (void)out_size; (void)ws_size;
  const float* Q  = (const float*)d_in[0];
  const float* K  = (const float*)d_in[1];
  const float* V  = (const float*)d_in[2];
  const float* Mg = (const float*)d_in[3];
  const float* zg = (const float*)d_in[4];

  float* outA = (float*)d_out;                       // [B,H,S,D]
  float* outM = outA + (long)NB * NH * NS * ND;      // [H,D,D]
  float* outz = outM + NH * ND * ND;                 // [H,D]

  float* dM_acc = (float*)d_ws;                      // [H,D,D] f32 accumulator
  float* z_acc  = dM_acc + NH * ND * ND;             // [H,D]

  const int nacc = NH * ND * ND + NH * ND;
  cm_zero_kernel<<<(nacc + 255) / 256, 256, 0, stream>>>(dM_acc, nacc);
  cm_main_kernel<<<NB * NH * (NS / SPW), 256, 0, stream>>>(
      Q, K, V, Mg, zg, outA, dM_acc, z_acc);
  cm_finalize_kernel<<<(nacc + 255) / 256, 256, 0, stream>>>(
      Mg, zg, dM_acc, z_acc, outM, outz);
}
